// Actor_53498112639267
// MI455X (gfx1250) — compile-verified, matrix operand order checked by reading, not run on hardware
//
#include <hip/hip_runtime.h>
#include <math.h>

#define N 4096
#define F 256
#define P 5
#define EPS 1e-8f

typedef _Float16 v16h __attribute__((ext_vector_type(16)));
typedef _Float16 v8h  __attribute__((ext_vector_type(8)));
typedef float    v8f  __attribute__((ext_vector_type(8)));
typedef float    v4f  __attribute__((ext_vector_type(4)));

// ---------------------------------------------------------------------------
// WMMA fragment loaders for v_wmma_f32_16x16x32_f16 (wave32).
// 16-bit A 16x32 layout (ISA 7.12.2): lane = m + 16*hi, hi in {0,1};
//   halfs 0..7  = K[k0 + 8*hi + 0..7]
//   halfs 8..15 = K[k0 + 16 + 8*hi + 0..7]
// B 32x16 layout mirrors this with lane <-> column n, so for a gram matrix
// (B = nf^T) the B fragment of a column block is loaded exactly like an A
// fragment of that row block. Two contiguous 16-byte loads per fragment.
// ---------------------------------------------------------------------------
__device__ __forceinline__ v16h load_frag_f16(const _Float16* __restrict__ base,
                                              int row0, int ld, int k0, int lane) {
  int m  = lane & 15;
  int hi = lane >> 4;
  const _Float16* rp = base + (size_t)(row0 + m) * ld + k0 + hi * 8;
  v8h lo = *(const v8h*)(rp);
  v8h up = *(const v8h*)(rp + 16);
  v16h r;
#pragma unroll
  for (int i = 0; i < 8; ++i) { r[i] = lo[i]; r[8 + i] = up[i]; }
  return r;
}

// Same fragment pattern but source is f32 (edges, values 0/1 -> exact in f16).
__device__ __forceinline__ v16h load_frag_f32cvt(const float* __restrict__ base,
                                                 int row0, int ld, int k0, int lane) {
  int m  = lane & 15;
  int hi = lane >> 4;
  const float* rp = base + (size_t)(row0 + m) * ld + k0 + hi * 8;
  v4f a0 = *(const v4f*)(rp);
  v4f a1 = *(const v4f*)(rp + 4);
  v4f b0 = *(const v4f*)(rp + 16);
  v4f b1 = *(const v4f*)(rp + 20);
  v16h r;
#pragma unroll
  for (int i = 0; i < 4; ++i) {
    r[i]      = (_Float16)a0[i];
    r[4 + i]  = (_Float16)a1[i];
    r[8 + i]  = (_Float16)b0[i];
    r[12 + i] = (_Float16)b1[i];
  }
  return r;
}

#define WMMA_F32_F16(A, B, C) \
  __builtin_amdgcn_wmma_f32_16x16x32_f16(false, (A), false, (B), (short)0, (C), false, false)

// ---------------------------------------------------------------------------
// K0: attributes [N][F] f32 -> At [F][N] f16 (transposed, so msg-GEMM B
// fragments are contiguous along K).
// ---------------------------------------------------------------------------
__global__ __launch_bounds__(256) void transpose_cvt_kernel(
    const float* __restrict__ attr, _Float16* __restrict__ At) {
  int idx = blockIdx.x * 256 + threadIdx.x;   // over F*N, coalesced writes
  if (idx < F * N) {
    int c = idx / N;
    int n = idx - c * N;
    At[idx] = (_Float16)attr[(size_t)n * F + c];
  }
}

// ---------------------------------------------------------------------------
// K1: gating arrays + per-persona scalar params.
// rowg[p][i] = persona[times][i][p] + (p==0)
// colg[p][j] = persona[times][j][p]
// pp[p] = {invT, scale} with scale = e / (e*exp(invT) + EPS)
//   (max of exp(S/T)*e is at S==1: diagonal of the gram of unit-norm rows)
// ---------------------------------------------------------------------------
__global__ __launch_bounds__(256) void prep_kernel(
    const float* __restrict__ persona, const float* __restrict__ T,
    const float* __restrict__ e, const int* __restrict__ times,
    float* __restrict__ rowg, float* __restrict__ colg, float* __restrict__ pp) {
  int n = blockIdx.x * 256 + threadIdx.x;
  int tm = times[0];
  if (n < N) {
#pragma unroll
    for (int p = 0; p < P; ++p) {
      float v = persona[((size_t)tm * N + n) * P + p];
      rowg[p * N + n] = v + (p == 0 ? 1.0f : 0.0f);
      colg[p * N + n] = v;
    }
  }
  if (blockIdx.x == 0 && threadIdx.x < P) {
    int p = threadIdx.x;
    float invT = 1.0f / (T[p] + EPS);
    float mx   = e[p] * __expf(invT);
    pp[2 * p + 0] = invT;
    pp[2 * p + 1] = e[p] / (mx + EPS);
  }
}

// ---------------------------------------------------------------------------
// K2: msg = edges @ attributes via WMMA.
// Block = 8 waves (4x2), wave owns 32x32 of msg[N][F]. K loop over 4096.
// ---------------------------------------------------------------------------
__global__ __launch_bounds__(256) void msg_gemm_kernel(
    const float* __restrict__ edges, const _Float16* __restrict__ At,
    float* __restrict__ msg) {
  int lane = threadIdx.x & 31;
  int wave = threadIdx.x >> 5;
  int wr = wave >> 1, wc = wave & 1;
  int row0 = blockIdx.y * 128 + wr * 32;
  int col0 = blockIdx.x * 64 + wc * 32;

  v8f c00 = {}, c01 = {}, c10 = {}, c11 = {};
  for (int k = 0; k < N; k += 32) {
    v16h a0 = load_frag_f32cvt(edges, row0,      N, k, lane);
    v16h a1 = load_frag_f32cvt(edges, row0 + 16, N, k, lane);
    v16h b0 = load_frag_f16(At, col0,      N, k, lane);
    v16h b1 = load_frag_f16(At, col0 + 16, N, k, lane);
    c00 = WMMA_F32_F16(a0, b0, c00);
    c01 = WMMA_F32_F16(a0, b1, c01);
    c10 = WMMA_F32_F16(a1, b0, c10);
    c11 = WMMA_F32_F16(a1, b1, c11);
  }
  int hi = lane >> 4, n = lane & 15;
#pragma unroll
  for (int v = 0; v < 8; ++v) {
    int i0 = row0 + hi * 8 + v;
    int i1 = i0 + 16;
    msg[(size_t)i0 * F + col0 + n]      = c00[v];
    msg[(size_t)i0 * F + col0 + 16 + n] = c01[v];
    msg[(size_t)i1 * F + col0 + n]      = c10[v];
    msg[(size_t)i1 * F + col0 + 16 + n] = c11[v];
  }
}

// ---------------------------------------------------------------------------
// K3: per-persona normalized features -> f16.
// nfh[p][n][k] = normalize_row(r*attr + (1-r)*W*msg)[k]
// One block per row n; 256 threads = F features; block reduction in LDS.
// ---------------------------------------------------------------------------
__global__ __launch_bounds__(256) void nf_kernel(
    const float* __restrict__ attr, const float* __restrict__ msg,
    const float* __restrict__ r, const float* __restrict__ W,
    _Float16* __restrict__ nfh) {
  int n = blockIdx.x;
  int t = threadIdx.x;
  __shared__ float red[256];
  float a = attr[(size_t)n * F + t];
  float m = msg[(size_t)n * F + t];
#pragma unroll
  for (int p = 0; p < P; ++p) {
    float rp = r[p], wp = W[p];
    float f = rp * a + (m * wp) * (1.0f - rp);
    red[t] = f * f;
    __syncthreads();
    for (int s = 128; s > 0; s >>= 1) {
      if (t < s) red[t] += red[t + s];
      __syncthreads();
    }
    float inv = 1.0f / sqrtf(red[0]);
    __syncthreads();
    nfh[((size_t)p * N + n) * F + t] = (_Float16)(f * inv);
  }
}

// ---------------------------------------------------------------------------
// K4 (hot loop): fused gram GEMM + epilogue.
// out[i][j] = sum_p tanh(scale_p * exp(s_p(i,j)*invT_p)) * colg_p[j] * rowg_p[i]
// Block = 8 waves (4x2) -> 128x64 tile; wave owns 32x32 (4 WMMA accumulators),
// personas in the outer loop so C registers are reused. nf (10.5 MB) lives in
// the 192 MB L2, so fragments are loaded straight from global.
// ---------------------------------------------------------------------------
__global__ __launch_bounds__(256) void gram_epilogue_kernel(
    const _Float16* __restrict__ nfh, const float* __restrict__ rowg,
    const float* __restrict__ colg, const float* __restrict__ pp,
    float* __restrict__ out) {
  int lane = threadIdx.x & 31;
  int wave = threadIdx.x >> 5;
  int wr = wave >> 1, wc = wave & 1;
  int row0 = blockIdx.y * 128 + wr * 32;
  int col0 = blockIdx.x * 64 + wc * 32;
  int hi = lane >> 4, nn = lane & 15;

  v8f acc00 = {}, acc01 = {}, acc10 = {}, acc11 = {};

#pragma unroll 1
  for (int p = 0; p < P; ++p) {
    const _Float16* nf = nfh + (size_t)p * N * F;
    v8f c00 = {}, c01 = {}, c10 = {}, c11 = {};
#pragma unroll
    for (int k = 0; k < F; k += 32) {
      v16h a0 = load_frag_f16(nf, row0,      F, k, lane);
      v16h a1 = load_frag_f16(nf, row0 + 16, F, k, lane);
      v16h b0 = load_frag_f16(nf, col0,      F, k, lane);
      v16h b1 = load_frag_f16(nf, col0 + 16, F, k, lane);
      c00 = WMMA_F32_F16(a0, b0, c00);
      c01 = WMMA_F32_F16(a0, b1, c01);
      c10 = WMMA_F32_F16(a1, b0, c10);
      c11 = WMMA_F32_F16(a1, b1, c11);
    }
    float invT  = pp[2 * p + 0];
    float scale = pp[2 * p + 1];
    float cg0 = colg[p * N + col0 + nn];
    float cg1 = colg[p * N + col0 + 16 + nn];
#pragma unroll
    for (int v = 0; v < 8; ++v) {
      int i0 = row0 + hi * 8 + v;
      float rg0 = rowg[p * N + i0];
      float rg1 = rowg[p * N + i0 + 16];
      acc00[v] += tanhf(scale * __expf(c00[v] * invT)) * cg0 * rg0;
      acc01[v] += tanhf(scale * __expf(c01[v] * invT)) * cg1 * rg0;
      acc10[v] += tanhf(scale * __expf(c10[v] * invT)) * cg0 * rg1;
      acc11[v] += tanhf(scale * __expf(c11[v] * invT)) * cg1 * rg1;
    }
  }

#pragma unroll
  for (int v = 0; v < 8; ++v) {
    int i0 = row0 + hi * 8 + v;
    int i1 = i0 + 16;
    out[(size_t)i0 * N + col0 + nn]      = acc00[v];
    out[(size_t)i0 * N + col0 + 16 + nn] = acc01[v];
    out[(size_t)i1 * N + col0 + nn]      = acc10[v];
    out[(size_t)i1 * N + col0 + 16 + nn] = acc11[v];
  }
}

// ---------------------------------------------------------------------------
extern "C" void kernel_launch(void* const* d_in, const int* in_sizes, int n_in,
                              void* d_out, int out_size, void* d_ws, size_t ws_size,
                              hipStream_t stream) {
  (void)in_sizes; (void)n_in; (void)out_size; (void)ws_size;
  const float* attributes = (const float*)d_in[0];
  const float* edges      = (const float*)d_in[1];
  const float* persona    = (const float*)d_in[2];
  const float* T          = (const float*)d_in[3];
  const float* e          = (const float*)d_in[4];
  const float* r          = (const float*)d_in[5];
  const float* W          = (const float*)d_in[6];
  const int*   times      = (const int*)d_in[7];
  float* out = (float*)d_out;

  char* ws = (char*)d_ws;
  float*     msg  = (float*)(ws);                                   // N*F f32   : 4 MB
  _Float16*  At   = (_Float16*)(ws + (size_t)N * F * 4);            // F*N f16   : 2 MB
  _Float16*  nfh  = (_Float16*)(ws + (size_t)N * F * 6);            // P*N*F f16 : 10.5 MB
  char* tail      = ws + (size_t)N * F * 6 + (size_t)P * N * F * 2;
  float*     rowg = (float*)(tail);                                 // P*N f32
  float*     colg = (float*)(tail + (size_t)P * N * 4);             // P*N f32
  float*     pp   = (float*)(tail + (size_t)2 * P * N * 4);         // P*2 f32

  transpose_cvt_kernel<<<dim3((N * F) / 256), dim3(256), 0, stream>>>(attributes, At);
  prep_kernel<<<dim3(N / 256), dim3(256), 0, stream>>>(persona, T, e, times, rowg, colg, pp);
  msg_gemm_kernel<<<dim3(F / 64, N / 128), dim3(256), 0, stream>>>(edges, At, msg);
  nf_kernel<<<dim3(N), dim3(256), 0, stream>>>(attributes, msg, r, W, nfh);
  gram_epilogue_kernel<<<dim3(N / 64, N / 128), dim3(256), 0, stream>>>(nfh, rowg, colg, pp, out);
}
